// MultiHeadCrossModalAttention_30236569764102
// MI455X (gfx1250) — compile-verified, matrix operand order checked by reading, not run on hardware
//
#include <hip/hip_runtime.h>
#include <math.h>

typedef __attribute__((ext_vector_type(16))) _Float16 v16h;
typedef __attribute__((ext_vector_type(8)))  float    v8f;
typedef __attribute__((ext_vector_type(4)))  unsigned int v4u;
typedef __attribute__((ext_vector_type(8)))  int      v8i_;
typedef __attribute__((ext_vector_type(4)))  int      v4i_;

static constexpr int Bn = 4, SQn = 1024, SKn = 2048, Dn = 768, Hn = 12, HDn = 64;

#if defined(__has_builtin)
#if __has_builtin(__builtin_amdgcn_tensor_load_to_lds) && __has_builtin(__builtin_amdgcn_s_wait_tensorcnt)
#define USE_TDM 1
#endif
#endif
#ifndef USE_TDM
#define USE_TDM 0
#endif

static __device__ __forceinline__ v8f wmma_f16(v16h a, v16h b, v8f c) {
    return __builtin_amdgcn_wmma_f32_16x16x32_f16(
        false, a, false, b, (short)0, c, false, false);
}

// ---------------------------------------------------------------------------
// f32 -> f16 straight convert (grid-stride)
// ---------------------------------------------------------------------------
__global__ __launch_bounds__(256)
void cvt_kernel(const float* __restrict__ in, _Float16* __restrict__ out, int n)
{
    for (size_t i = (size_t)blockIdx.x * 256 + threadIdx.x; i < (size_t)n;
         i += (size_t)gridDim.x * 256)
        out[i] = (_Float16)in[i];
}

// ---------------------------------------------------------------------------
// W[768,768] f32 row-major -> WT[768,768] f16 with WT[n][k] = W[k][n]
// ---------------------------------------------------------------------------
__global__ __launch_bounds__(256)
void cvtT_kernel(const float* __restrict__ W, _Float16* __restrict__ WT)
{
    const size_t idx = (size_t)blockIdx.x * 256 + threadIdx.x;   // over 768*768
    const int k = (int)(idx / Dn);
    const int n = (int)(idx % Dn);
    WT[(size_t)n * Dn + k] = (_Float16)W[idx];
}

// ---------------------------------------------------------------------------
// WMMA GEMM: out = A16[M,768] @ WT^T  (WT is f16, [n][k] layout).
// Both fragments are contiguous 32B loads along K.
// mode 0: -> Qh [b,h,q,64] f16, scaled by 0.125/temperature
// mode 1: -> Kh [b,h,k,64] f16
// mode 2: -> Vt [b,h,64,sk] f16 (transposed for PV B-fragments)
// mode 3: -> xbuf f32 = acc + bo + query (residual)
// ---------------------------------------------------------------------------
__global__ __launch_bounds__(128)
void proj_gemm_kernel(const _Float16* __restrict__ A16, const _Float16* __restrict__ WT,
                      const float* __restrict__ bias, const float* __restrict__ resid,
                      const float* __restrict__ temp,
                      _Float16* __restrict__ out16, float* __restrict__ out32,
                      int S, int mode)
{
    const int lane  = threadIdx.x & 31;
    const int wave  = threadIdx.x >> 5;
    const int nBase = (blockIdx.x * 4 + wave) * 16;
    const int mBase = blockIdx.y * 16;
    const int lm    = lane & 15;
    const int koff  = (lane >> 4) * 16;

    const _Float16* pa = A16 + (size_t)(mBase + lm) * Dn + koff;
    const _Float16* pb = WT  + (size_t)(nBase + lm) * Dn + koff;

    v8f acc = {};
    for (int kk = 0; kk < Dn; kk += 32) {
        __builtin_prefetch(pa + kk + 64, 0, 3);
        __builtin_prefetch(pb + kk + 64, 0, 3);
        const v16h a  = *(const v16h*)(pa + kk);
        const v16h bf = *(const v16h*)(pb + kk);
        acc = wmma_f16(a, bf, acc);
    }

    float qs = 1.0f;
    if (mode == 0) qs = 0.125f / temp[0];   // HD^-0.5 / temperature folded into Q

    #pragma unroll
    for (int r = 0; r < 8; ++r) {
        const int m = mBase + r + (lane >> 4) * 8;
        const int n = nBase + lm;
        const float v = acc[r] * qs;
        const int bb = m / S, s = m % S;
        if (mode == 0) {
            const int h = n >> 6, hd = n & 63;
            out16[(((size_t)bb * Hn + h) * SQn + s) * HDn + hd] = (_Float16)v;
        } else if (mode == 1) {
            const int h = n >> 6, hd = n & 63;
            out16[(((size_t)bb * Hn + h) * SKn + s) * HDn + hd] = (_Float16)v;
        } else if (mode == 2) {
            const int h = n >> 6, hd = n & 63;
            out16[(((size_t)bb * Hn + h) * HDn + hd) * SKn + s] = (_Float16)v;
        } else {
            out32[(size_t)m * Dn + n] = v + bias[n] + resid[(size_t)m * Dn + n];
        }
    }
}

// ---------------------------------------------------------------------------
// Fused attention: one block per (b*h, 16-row q tile). 256 threads = 8 waves.
// LDS: 16x2048 f32 score strip (128KB) + 8x256 partials (8KB) + 16x64 f16
// Q tile (2KB, staged by the Tensor Data Mover). attn hits HBM exactly once.
// ---------------------------------------------------------------------------
static constexpr unsigned QTILE_OFF_BYTES = (unsigned)((16 * SKn + 8 * 256) * 4);

__global__ __launch_bounds__(256)
void attn_kernel(const _Float16* __restrict__ Qh, const _Float16* __restrict__ Kh,
                 const _Float16* __restrict__ Vt, const int* __restrict__ mask,
                 _Float16* __restrict__ ctx16, float* __restrict__ attn)
{
    extern __shared__ float smem[];
    float*    sc    = smem;                        // [16][SKn]
    float*    part  = smem + 16 * SKn;             // [8][256]
    _Float16* qtile = (_Float16*)((char*)smem + QTILE_OFF_BYTES);  // [16][64]

    const int bh    = blockIdx.x;                  // b*H + h
    const int bb    = bh / Hn;
    const int h     = bh % Hn;
    const int qBase = blockIdx.y * 16;
    const int lane  = threadIdx.x & 31;
    const int wave  = threadIdx.x >> 5;
    const int lm    = lane & 15;
    const int koff  = (lane >> 4) * 16;

    const _Float16* Qbh = Qh + (size_t)bh * SQn * HDn;
    const _Float16* Kbh = Kh + (size_t)bh * SKn * HDn;
    const _Float16* Vbh = Vt + (size_t)bh * HDn * SKn;
    const int*      mk  = mask + (size_t)bb * SKn;

    // ---- stage Q tile (16x64 f16 = 2KB) into LDS via TDM ----
#if USE_TDM
    if (wave == 0) {
        const unsigned long long gaddr =
            (unsigned long long)(const void*)(Qbh + (size_t)qBase * HDn);
        v4u g0;
        g0[0] = 1u;                                    // count = 1 valid descriptor
        g0[1] = QTILE_OFF_BYTES;                       // lds_addr
        g0[2] = (unsigned)(gaddr & 0xffffffffu);       // global_addr[31:0]
        g0[3] = (unsigned)((gaddr >> 32) & 0x01ffffffu) | (2u << 30); // addr[56:32] | type=2
        v8i_ g1;
        g1[0] = (int)(1u << 16);                       // data_size = 2 bytes
        g1[1] = (int)(64u << 16);                      // tensor_dim0 = 64
        g1[2] = (int)(16u << 16);                      // tensor_dim1 = 16
        g1[3] = (int)(64u << 16);                      // tile_dim0 = 64
        g1[4] = 16;                                    // tile_dim1 = 16
        g1[5] = 64;                                    // tensor_dim0_stride = 64
        g1[6] = 0;
        g1[7] = 0;
        v4i_ gz = {0, 0, 0, 0};
#if __clang_major__ >= 23
        v8i_ gz8 = {0, 0, 0, 0, 0, 0, 0, 0};
        __builtin_amdgcn_tensor_load_to_lds(g0, g1, gz, gz, gz8, 0);
#else
        __builtin_amdgcn_tensor_load_to_lds(g0, g1, gz, gz, 0);
#endif
        __builtin_amdgcn_s_wait_tensorcnt(0);
    }
#else
    for (int i = threadIdx.x; i < 16 * HDn; i += 256)
        qtile[i] = Qbh[(size_t)qBase * HDn + i];
#endif
    __syncthreads();

    // Q fragments from LDS (held in registers for the whole k sweep)
    const v16h qa0 = *(const v16h*)(qtile + lm * HDn + 0  + koff);
    const v16h qa1 = *(const v16h*)(qtile + lm * HDn + 32 + koff);

    // ---- phase 1: scores = Q K^T (scale pre-folded), masked, into LDS ----
    for (int kt = wave; kt < SKn / 16; kt += 8) {
        const v16h kb0 = *(const v16h*)(Kbh + (size_t)(kt * 16 + lm) * HDn + 0  + koff);
        const v16h kb1 = *(const v16h*)(Kbh + (size_t)(kt * 16 + lm) * HDn + 32 + koff);
        v8f acc = {};
        acc = wmma_f16(qa0, kb0, acc);
        acc = wmma_f16(qa1, kb1, acc);
        const int kg = kt * 16 + lm;
        const bool masked = (mk[kg] == 0);
        #pragma unroll
        for (int r = 0; r < 8; ++r) {
            const int m = r + (lane >> 4) * 8;
            sc[m * SKn + kg] = masked ? -__builtin_inff() : acc[r];
        }
    }
    __syncthreads();

    // ---- phase 2: row softmax (wave handles 2 rows), write attn to HBM ----
    for (int i = 0; i < 2; ++i) {
        const int row = wave * 2 + i;
        float* rp = sc + row * SKn;
        float mx = -__builtin_inff();
        for (int k = lane; k < SKn; k += 32) mx = fmaxf(mx, rp[k]);
        for (int off = 16; off > 0; off >>= 1) mx = fmaxf(mx, __shfl_xor(mx, off));
        float sum = 0.f;
        for (int k = lane; k < SKn; k += 32) { float p = __expf(rp[k] - mx); rp[k] = p; sum += p; }
        for (int off = 16; off > 0; off >>= 1) sum += __shfl_xor(sum, off);
        const float inv = 1.0f / sum;
        float* arow = attn + ((size_t)bh * SQn + qBase + row) * SKn;
        for (int k = lane; k < SKn; k += 32) { float p = rp[k] * inv; rp[k] = p; arow[k] = p; }
    }
    __syncthreads();

    // ---- phase 3: ctx = P @ V, K-dim split across wave pairs ----
    const int t    = wave & 3;   // 16-wide hd tile
    const int half = wave >> 2;  // k half
    v8f acc = {};
    for (int kk = half * (SKn / 2); kk < (half + 1) * (SKn / 2); kk += 32) {
        v16h a;
        const float* pa = sc + lm * SKn + kk + koff;
        #pragma unroll
        for (int j = 0; j < 16; ++j) a[j] = (_Float16)pa[j];
        const v16h vb = *(const v16h*)(Vbh + (size_t)(t * 16 + lm) * SKn + kk + koff);
        acc = wmma_f16(a, vb, acc);
    }
    #pragma unroll
    for (int r = 0; r < 8; ++r) {
        const int m = r + (lane >> 4) * 8;
        part[wave * 256 + m * 16 + lm] = acc[r];
    }
    __syncthreads();

    for (int e = threadIdx.x; e < 1024; e += 256) {
        const int tt = e >> 8, idx = e & 255;
        const int m = idx >> 4, n = idx & 15;
        const float v = part[tt * 256 + idx] + part[(tt + 4) * 256 + idx];
        ctx16[((size_t)bb * SQn + qBase + m) * Dn + h * 64 + tt * 16 + n] = (_Float16)v;
    }
}

// ---------------------------------------------------------------------------
// Per-row LayerNorm: one block per row of x[4096, 768]
// ---------------------------------------------------------------------------
__global__ __launch_bounds__(256)
void ln_kernel(const float* __restrict__ xbuf, const float* __restrict__ gamma,
               const float* __restrict__ beta, float* __restrict__ out)
{
    __shared__ float red[256];
    const int row = blockIdx.x;
    const float* x = xbuf + (size_t)row * Dn;

    float s = 0.f;
    for (int i = threadIdx.x; i < Dn; i += 256) s += x[i];
    red[threadIdx.x] = s; __syncthreads();
    for (int off = 128; off > 0; off >>= 1) {
        if (threadIdx.x < off) red[threadIdx.x] += red[threadIdx.x + off];
        __syncthreads();
    }
    const float mu = red[0] / (float)Dn;
    __syncthreads();

    float v = 0.f;
    for (int i = threadIdx.x; i < Dn; i += 256) { float d = x[i] - mu; v += d * d; }
    red[threadIdx.x] = v; __syncthreads();
    for (int off = 128; off > 0; off >>= 1) {
        if (threadIdx.x < off) red[threadIdx.x] += red[threadIdx.x + off];
        __syncthreads();
    }
    const float rstd = rsqrtf(red[0] / (float)Dn + 1e-5f);

    for (int i = threadIdx.x; i < Dn; i += 256)
        out[(size_t)row * Dn + i] = gamma[i] * (x[i] - mu) * rstd + beta[i];
}

// ---------------------------------------------------------------------------
extern "C" void kernel_launch(void* const* d_in, const int* in_sizes, int n_in,
                              void* d_out, int out_size, void* d_ws, size_t ws_size,
                              hipStream_t stream) {
    const float* query = (const float*)d_in[0];
    const float* key   = (const float*)d_in[1];
    const float* value = (const float*)d_in[2];
    const int*   mask  = (const int*)  d_in[3];
    const float* Wq    = (const float*)d_in[4];
    const float* Wk    = (const float*)d_in[5];
    const float* Wv    = (const float*)d_in[6];
    const float* Wo    = (const float*)d_in[7];
    const float* bo    = (const float*)d_in[8];
    const float* gamma = (const float*)d_in[9];
    const float* beta  = (const float*)d_in[10];
    const float* temp  = (const float*)d_in[11];

    float* out  = (float*)d_out;
    float* attn = out + (size_t)Bn * SQn * Dn;   // out then attn, flat

    // workspace layout (byte offsets)
    char* ws = (char*)d_ws;
    _Float16* Qh   = (_Float16*)(ws);              //  6291456
    _Float16* Kh   = (_Float16*)(ws +  6291456);   // 12582912
    _Float16* Vt   = (_Float16*)(ws + 18874368);   // 12582912
    _Float16* ctxh = (_Float16*)(ws + 31457280);   //  6291456
    _Float16* qh32 = (_Float16*)(ws + 37748736);   //  6291456  query f16
    _Float16* kh32 = (_Float16*)(ws + 44040192);   // 12582912  key   f16
    _Float16* vh32 = (_Float16*)(ws + 56623104);   // 12582912  value f16
    _Float16* WqT  = (_Float16*)(ws + 69206016);   //  1179648
    _Float16* WkT  = (_Float16*)(ws + 70385664);
    _Float16* WvT  = (_Float16*)(ws + 71565312);
    _Float16* WoT  = (_Float16*)(ws + 72744960);
    float*    xbuf = (float*)(ws + 37748736);      // overlaps qh32/kh32 (dead by then)

    const int nq = Bn * SQn * Dn, nk = Bn * SKn * Dn, nw = Dn * Dn;

    // one-time f16 pre-pass: activations straight, weights transposed
    cvt_kernel <<<dim3((nq + 255) / 256), dim3(256), 0, stream>>>(query, qh32, nq);
    cvt_kernel <<<dim3((nk + 255) / 256), dim3(256), 0, stream>>>(key,   kh32, nk);
    cvt_kernel <<<dim3((nk + 255) / 256), dim3(256), 0, stream>>>(value, vh32, nk);
    cvtT_kernel<<<dim3(nw / 256), dim3(256), 0, stream>>>(Wq, WqT);
    cvtT_kernel<<<dim3(nw / 256), dim3(256), 0, stream>>>(Wk, WkT);
    cvtT_kernel<<<dim3(nw / 256), dim3(256), 0, stream>>>(Wv, WvT);
    cvtT_kernel<<<dim3(nw / 256), dim3(256), 0, stream>>>(Wo, WoT);

    const dim3 gblk(128);
    proj_gemm_kernel<<<dim3(Dn / 64, (Bn * SQn) / 16), gblk, 0, stream>>>(
        qh32, WqT, nullptr, nullptr, temp, Qh, nullptr, SQn, 0);
    proj_gemm_kernel<<<dim3(Dn / 64, (Bn * SKn) / 16), gblk, 0, stream>>>(
        kh32, WkT, nullptr, nullptr, temp, Kh, nullptr, SKn, 1);
    proj_gemm_kernel<<<dim3(Dn / 64, (Bn * SKn) / 16), gblk, 0, stream>>>(
        vh32, WvT, nullptr, nullptr, temp, Vt, nullptr, SKn, 2);

    const size_t smem_bytes = QTILE_OFF_BYTES + 16 * HDn * sizeof(_Float16); // ~138 KB
    attn_kernel<<<dim3(Bn * Hn, SQn / 16), dim3(256), smem_bytes, stream>>>(
        Qh, Kh, Vt, mask, ctxh, attn);

    proj_gemm_kernel<<<dim3(Dn / 64, (Bn * SQn) / 16), gblk, 0, stream>>>(
        ctxh, WoT, bo, query, nullptr, nullptr, xbuf, SQn, 3);

    ln_kernel<<<dim3(Bn * SQn), dim3(256), 0, stream>>>(xbuf, gamma, beta, out);
}
